// ColorizationLoss_72215580115022
// MI455X (gfx1250) — matrix operand order repaired and checked by reading, not compiled, not run on hardware
//
#include <hip/hip_runtime.h>
#include <hip/hip_bf16.h>
#include <math.h>

typedef _Float16 v16h __attribute__((ext_vector_type(16)));
typedef _Float16 v8h  __attribute__((ext_vector_type(8)));
typedef float    v8f  __attribute__((ext_vector_type(8)));

#define BATCH 4
#define H0 512
#define W0 512
#define HW0 (H0 * W0)
#define H1 256
#define W1 256
#define HW1 (H1 * W1)
// padded spatial dims (1-pixel zero halo)
#define W0P (W0 + 2)
#define H0P (H0 + 2)
#define W1P (W1 + 2)
#define H1P (H1 + 2)

// ---------------------------------------------------------------------------
// small helpers
// ---------------------------------------------------------------------------
__device__ __forceinline__ void warp_atomic_add(float v, float* dst) {
    #pragma unroll
    for (int off = 16; off > 0; off >>= 1)
        v += __shfl_down(v, off, 32);
    if ((threadIdx.x & 31) == 0) atomicAdd(dst, v);
}

__device__ __forceinline__ float lab_finv(float t) {
    const float delta = 6.0f / 29.0f;
    return (t > delta) ? t * t * t : 3.0f * delta * delta * (t - 4.0f / 29.0f);
}

__device__ __forceinline__ float srgb_gamma(float t) {
    t = fminf(fmaxf(t, 0.0f), 1.0f);
    float g = (t <= 0.0031308f) ? 12.92f * t
                                : 1.055f * powf(fmaxf(t, 1e-8f), 1.0f / 2.4f) - 0.055f;
    return fminf(fmaxf(g, 0.0f), 1.0f);
}

__device__ __forceinline__ void lab2rgb(float a, float b, float& R, float& G, float& B) {
    float fy = 16.0f / 116.0f;                 // L == 0
    float fx = fy + (a * 128.0f) / 500.0f;
    float fz = fy - (b * 128.0f) / 200.0f;
    float X = 0.950456f * lab_finv(fx);
    float Y = lab_finv(fy);
    float Z = 1.088754f * lab_finv(fz);
    R = srgb_gamma( 3.240479f * X - 1.537150f * Y - 0.498535f * Z);
    G = srgb_gamma(-0.969256f * X + 1.875992f * Y + 0.041556f * Z);
    B = srgb_gamma( 0.055648f * X - 0.196174f * Y + 1.057311f * Z);
}

// ---------------------------------------------------------------------------
// accumulator init + buffer zeroing
// ---------------------------------------------------------------------------
__global__ void init_acc_kernel(float* acc) {
    if (threadIdx.x < 3) acc[threadIdx.x] = 0.0f;
}

// zero n16 128-bit chunks
__global__ void zero_b128_kernel(uint4* __restrict__ p, int n16) {
    int i = blockIdx.x * blockDim.x + threadIdx.x;
    uint4 z = {0, 0, 0, 0};
    if (i < n16) p[i] = z;
}

// ---------------------------------------------------------------------------
// pack f32 OIHW weights -> f16 GEMM panel [K][O],  k = rs*Cin + c  (tap-major)
// ---------------------------------------------------------------------------
__global__ void pack_weights_kernel(const float* __restrict__ w, _Float16* __restrict__ panel,
                                    int O, int I) {
    int idx = blockIdx.x * blockDim.x + threadIdx.x;
    int Ktot = I * 9;
    if (idx >= Ktot * O) return;
    int k = idx / O;
    int o = idx - k * O;
    int rs = k / I;
    int c  = k - rs * I;
    panel[k * O + o] = (_Float16)w[(o * I + c) * 9 + rs];
}

// ---------------------------------------------------------------------------
// fused LAB->RGB (both streams) + MSE partial sum. rgb out is planar NCHW f16.
// ---------------------------------------------------------------------------
__global__ void lab2rgb_mse_kernel(const float* __restrict__ pred, const float* __restrict__ targ,
                                   _Float16* __restrict__ rgbP, _Float16* __restrict__ rgbT,
                                   float* acc) {
    int idx = blockIdx.x * blockDim.x + threadIdx.x;       // over BATCH*HW0
    int b = idx / HW0;
    int p = idx - b * HW0;
    float pa = pred[(b * 2 + 0) * HW0 + p];
    float pb = pred[(b * 2 + 1) * HW0 + p];
    float ta = targ[(b * 2 + 0) * HW0 + p];
    float tb = targ[(b * 2 + 1) * HW0 + p];

    float r, g, bl;
    lab2rgb(pa, pb, r, g, bl);
    rgbP[(b * 3 + 0) * HW0 + p] = (_Float16)r;
    rgbP[(b * 3 + 1) * HW0 + p] = (_Float16)g;
    rgbP[(b * 3 + 2) * HW0 + p] = (_Float16)bl;
    lab2rgb(ta, tb, r, g, bl);
    rgbT[(b * 3 + 0) * HW0 + p] = (_Float16)r;
    rgbT[(b * 3 + 1) * HW0 + p] = (_Float16)g;
    rgbT[(b * 3 + 2) * HW0 + p] = (_Float16)bl;

    float da = pa - ta, db = pb - tb;
    warp_atomic_add(da * da + db * db, &acc[0]);
}

// ---------------------------------------------------------------------------
// SSIM horizontal pass: 5 moment maps (sum over 11 x-taps, zero pad)
// hbuf layout: [(b*10 + map*2 + ch)][HW0]  f32
// ---------------------------------------------------------------------------
__global__ void ssim_h_kernel(const float* __restrict__ pred, const float* __restrict__ targ,
                              float* __restrict__ hbuf) {
    int idx = blockIdx.x * blockDim.x + threadIdx.x;       // over BATCH*2*HW0
    int b  = idx / (2 * HW0);
    int r  = idx - b * 2 * HW0;
    int ch = r / HW0;
    int p  = r - ch * HW0;
    int y = p / W0, x = p - y * W0;
    const float* P = pred + (b * 2 + ch) * HW0 + y * W0;
    const float* T = targ + (b * 2 + ch) * HW0 + y * W0;
    float sp = 0, st = 0, spp = 0, stt = 0, spt = 0;
    #pragma unroll
    for (int dx = -5; dx <= 5; ++dx) {
        int xx = x + dx;
        if (xx >= 0 && xx < W0) {
            float pv = P[xx], tv = T[xx];
            sp += pv; st += tv; spp += pv * pv; stt += tv * tv; spt += pv * tv;
        }
    }
    size_t base = (size_t)(b * 10 + ch) * HW0 + p;
    hbuf[base + (size_t)0 * 2 * HW0] = sp;
    hbuf[base + (size_t)1 * 2 * HW0] = st;
    hbuf[base + (size_t)2 * 2 * HW0] = spp;
    hbuf[base + (size_t)3 * 2 * HW0] = stt;
    hbuf[base + (size_t)4 * 2 * HW0] = spt;
}

// ---------------------------------------------------------------------------
// SSIM vertical pass + map evaluation + reduction
// ---------------------------------------------------------------------------
__global__ void ssim_v_kernel(const float* __restrict__ hbuf, float* acc) {
    int idx = blockIdx.x * blockDim.x + threadIdx.x;       // over BATCH*2*HW0
    int b  = idx / (2 * HW0);
    int r  = idx - b * 2 * HW0;
    int ch = r / HW0;
    int p  = r - ch * HW0;
    int y = p / W0, x = p - y * W0;
    const float* base = hbuf + (size_t)(b * 10 + ch) * HW0;
    float S[5] = {0, 0, 0, 0, 0};
    for (int dy = -5; dy <= 5; ++dy) {
        int yy = y + dy;
        if (yy >= 0 && yy < H0) {
            size_t off = (size_t)yy * W0 + x;
            #pragma unroll
            for (int m = 0; m < 5; ++m)
                S[m] += base[(size_t)m * 2 * HW0 + off];
        }
    }
    const float inv = 1.0f / 121.0f;
    float mu1 = S[0] * inv, mu2 = S[1] * inv;
    float m11 = S[2] * inv, m22 = S[3] * inv, m12 = S[4] * inv;
    float s1 = m11 - mu1 * mu1;
    float s2 = m22 - mu2 * mu2;
    float s12 = m12 - mu1 * mu2;
    const float C1 = 0.0001f, C2 = 0.0009f;
    float ssim = ((2.0f * mu1 * mu2 + C1) * (2.0f * s12 + C2)) /
                 ((mu1 * mu1 + mu2 * mu2 + C1) * (s1 + s2 + C2));
    warp_atomic_add(ssim, &acc[2]);
}

// ---------------------------------------------------------------------------
// conv1: 3 -> 64 @512x512, relu, direct VALU conv (~2% of FLOPs).
// rgb is planar NCHW; output written into padded NHWC interior.
// ---------------------------------------------------------------------------
__global__ void conv1_direct_kernel(const _Float16* __restrict__ rgb,
                                    const float* __restrict__ w, const float* __restrict__ bias,
                                    _Float16* __restrict__ out) {
    __shared__ float wl[64 * 27];
    __shared__ float bl[64];
    for (int i = threadIdx.x; i < 64 * 27; i += 256) wl[i] = w[i];
    if (threadIdx.x < 64) bl[threadIdx.x] = bias[threadIdx.x];
    __syncthreads();

    const int o = threadIdx.x & 63;
    const int q = threadIdx.x >> 6;                        // 0..3
    const int p = blockIdx.x * 4 + q;                      // pixel in image
    const int b = blockIdx.y;
    const int y = p / W0, x = p - y * W0;
    const _Float16* in = rgb + (size_t)b * 3 * HW0;
    float s = bl[o];
    #pragma unroll
    for (int c = 0; c < 3; ++c) {
        #pragma unroll
        for (int rr = 0; rr < 3; ++rr) {
            int hh = y + rr - 1;
            #pragma unroll
            for (int ss = 0; ss < 3; ++ss) {
                int ww = x + ss - 1;
                float v = (hh >= 0 && hh < H0 && ww >= 0 && ww < W0)
                          ? (float)in[(c * H0 + hh) * W0 + ww] : 0.0f;
                s += v * wl[o * 27 + c * 9 + rr * 3 + ss];
            }
        }
    }
    out[((size_t)b * H0P * W0P + (size_t)(y + 1) * W0P + (x + 1)) * 64 + o] =
        (_Float16)fmaxf(s, 0.0f);
}

// ---------------------------------------------------------------------------
// Implicit-GEMM 3x3 conv via V_WMMA_F32_16X16X32_F16, CIN in {64,128}.
//  * activations NHWC with 1-pixel zero halo -> NO bounds checks.
//  * K ordering k = rs*CIN + c: each 32-wide chunk = one tap + 32 channels;
//    a lane's A fragment = two contiguous 16B runs -> 2x global b128 loads.
//  * 2x2 register blocking: each wave owns 2 M-tiles x 2 N-tiles; per chunk
//    2 A loads + 2 B ds loads feed 4 WMMAs (half the bytes/FLOP).
//  * block = 256 thr = 8 waves -> 256 consecutive positions (one image row,
//    W % 256 == 0 -> scalar h, w0) x 32 output channels.
//  in  : pointer to padded-(0,0) origin, row stride inW=W+2, batch inBS
//  out : pointer to output (0,0) position, row stride outW, batch outBS
// ---------------------------------------------------------------------------
template <int CIN>
__global__ void __launch_bounds__(256)
conv3x3_wmma_kernel(const _Float16* __restrict__ in, const _Float16* __restrict__ wpanel,
                    const float* __restrict__ bias, _Float16* __restrict__ out,
                    int Cout, int W, int inW, int outW, long inBS, long outBS, int relu) {
    constexpr int CPR = CIN >> 5;          // K-chunks per tap
    constexpr int NCHUNK = 9 * CPR;
    // B panel for 32 output channels (2 N-tiles), fragment order:
    // lds_b[kc][nt][lane][e]
    __shared__ __align__(16) _Float16 lds_b[NCHUNK * 1024];

    const int tid  = threadIdx.x;
    const int wave = tid >> 5;
    const int lane = tid & 31;
    const int lane15 = lane & 15;
    const int g    = lane >> 4;            // half-wave group
    const int nbase = blockIdx.y * 32;

    for (int i = tid; i < NCHUNK * 1024; i += 256) {
        int e  = i & 15;
        int l  = (i >> 4) & 31;
        int nt = (i >> 9) & 1;
        int kc = i >> 10;
        int k  = kc * 32 + ((e >> 3) << 4) + ((l >> 4) << 3) + (e & 7);
        lds_b[i] = wpanel[k * Cout + nbase + nt * 16 + (l & 15)];
    }
    __syncthreads();

    const int m0 = (blockIdx.x * 8 + wave) * 32;     // wave covers 32 positions
    const int h  = m0 / W;                           // scalar: one image row
    const int w0 = m0 - h * W;
    const _Float16* inB = in + (size_t)blockIdx.z * inBS;
    const int aLane = lane15 * CIN + g * 8;

    v8f acc00 = {}, acc01 = {}, acc10 = {}, acc11 = {};
    #pragma unroll
    for (int rr = 0; rr < 3; ++rr) {
        #pragma unroll
        for (int ss = 0; ss < 3; ++ss) {
            const int rsBase = ((h + rr) * inW + (w0 + ss)) * CIN + aLane;
            #pragma unroll
            for (int cc = 0; cc < CPR; ++cc) {
                const int kc = (rr * 3 + ss) * CPR + cc;
                const _Float16* ap = inB + rsBase + cc * 32;
                v16h a0, a1;
                {
                    v8h* p0 = (v8h*)&a0;
                    p0[0] = *(const v8h*)(ap);                    // ch +0..7
                    p0[1] = *(const v8h*)(ap + 16);               // ch +16..23
                    v8h* p1 = (v8h*)&a1;
                    p1[0] = *(const v8h*)(ap + 16 * CIN);         // M-tile 1
                    p1[1] = *(const v8h*)(ap + 16 * CIN + 16);
                }
                v16h b0 = *(const v16h*)(&lds_b[kc * 1024 + lane * 16]);
                v16h b1 = *(const v16h*)(&lds_b[kc * 1024 + 512 + lane * 16]);
                acc00 = __builtin_amdgcn_wmma_f32_16x16x32_f16(false, a0, false, b0, (short)0, acc00, false, false);
                acc01 = __builtin_amdgcn_wmma_f32_16x16x32_f16(false, a0, false, b1, (short)0, acc01, false, false);
                acc10 = __builtin_amdgcn_wmma_f32_16x16x32_f16(false, a1, false, b0, (short)0, acc10, false, false);
                acc11 = __builtin_amdgcn_wmma_f32_16x16x32_f16(false, a1, false, b1, (short)0, acc11, false, false);
            }
        }
    }

    _Float16* outB = out + (size_t)blockIdx.z * outBS;
    const v8f* accs[4] = { &acc00, &acc01, &acc10, &acc11 };
    #pragma unroll
    for (int mt = 0; mt < 2; ++mt) {
        #pragma unroll
        for (int nt = 0; nt < 2; ++nt) {
            const v8f& a = *accs[mt * 2 + nt];
            const int n = nbase + nt * 16 + lane15;
            const float bv = bias[n];
            #pragma unroll
            for (int r = 0; r < 8; ++r) {
                int wpos = w0 + mt * 16 + r + 8 * g;
                float v = a[r] + bv;
                if (relu) v = fmaxf(v, 0.0f);
                outB[(size_t)(h * outW + wpos) * Cout + n] = (_Float16)v;
            }
        }
    }
}

// ---------------------------------------------------------------------------
// 2x2 maxpool stride 2, NHWC padded (514x514x64) -> NHWC padded (258x258x64)
// ---------------------------------------------------------------------------
__global__ void maxpool_kernel(const _Float16* __restrict__ in, _Float16* __restrict__ out) {
    int idx = blockIdx.x * blockDim.x + threadIdx.x;       // over BATCH*HW1*64
    int c  = idx & 63;
    int t  = idx >> 6;
    int b  = t / HW1;
    int p  = t - b * HW1;
    int y = p / W1, x = p - y * W1;
    const _Float16* src = in + ((size_t)b * H0P * W0P + (size_t)(2 * y + 1) * W0P + (2 * x + 1)) * 64 + c;
    float a0 = (float)src[0];
    float a1 = (float)src[64];
    float a2 = (float)src[(size_t)W0P * 64];
    float a3 = (float)src[(size_t)W0P * 64 + 64];
    out[((size_t)b * H1P * W1P + (size_t)(y + 1) * W1P + (x + 1)) * 64 + c] =
        (_Float16)fmaxf(fmaxf(a0, a1), fmaxf(a2, a3));
}

// ---------------------------------------------------------------------------
// perceptual reduction: sum((pf - tf)^2)
// ---------------------------------------------------------------------------
__global__ void perc_reduce_kernel(const _Float16* __restrict__ a, const _Float16* __restrict__ b,
                                   float* acc, int n) {
    float s = 0.0f;
    for (int i = blockIdx.x * blockDim.x + threadIdx.x; i < n; i += gridDim.x * blockDim.x) {
        float d = (float)a[i] - (float)b[i];
        s += d * d;
    }
    warp_atomic_add(s, &acc[1]);
}

// ---------------------------------------------------------------------------
// final combine
// ---------------------------------------------------------------------------
__global__ void combine_kernel(const float* acc, float* out) {
    if (threadIdx.x == 0 && blockIdx.x == 0) {
        const float nPix  = (float)BATCH * 2.0f * (float)HW0;
        const float nPerc = (float)BATCH * 128.0f * (float)HW1;
        float mse  = acc[0] / nPix;
        float perc = acc[1] / nPerc;
        float ssim_mean = acc[2] / nPix;
        out[0] = mse + 0.2f * perc + 0.2f * (1.0f - ssim_mean);
    }
}

// ---------------------------------------------------------------------------
// launcher
// ---------------------------------------------------------------------------
extern "C" void kernel_launch(void* const* d_in, const int* in_sizes, int n_in,
                              void* d_out, int out_size, void* d_ws, size_t ws_size,
                              hipStream_t stream) {
    (void)in_sizes; (void)n_in; (void)out_size; (void)ws_size;

    const float* pred = (const float*)d_in[0];
    const float* targ = (const float*)d_in[1];
    const float* w1 = (const float*)d_in[2]; const float* b1 = (const float*)d_in[3];
    const float* w2 = (const float*)d_in[4]; const float* b2 = (const float*)d_in[5];
    const float* w3 = (const float*)d_in[6]; const float* b3 = (const float*)d_in[7];
    const float* w4 = (const float*)d_in[8]; const float* b4 = (const float*)d_in[9];
    float* out = (float*)d_out;

    // padded NHWC buffer sizes (elements, all batches)
    const size_t A1P = (size_t)BATCH * H0P * W0P * 64;    // conv1/conv2 act @512
    const size_t PLP = (size_t)BATCH * H1P * W1P * 64;    // pooled act @256
    const size_t A3P = (size_t)BATCH * H1P * W1P * 128;   // conv3 act @256
    const size_t FNP = (size_t)BATCH * HW1 * 128;         // conv4 out (no halo)

    char* ws = (char*)d_ws;
    size_t off = 0;
    auto alloc = [&](size_t sz) { size_t o = off; off = (off + sz + 255) & ~(size_t)255; return o; };

    float*     acc   = (float*)    (ws + alloc(16));
    _Float16*  p2    = (_Float16*) (ws + alloc((size_t)576  * 64  * 2));
    _Float16*  p3    = (_Float16*) (ws + alloc((size_t)576  * 128 * 2));
    _Float16*  p4    = (_Float16*) (ws + alloc((size_t)1152 * 128 * 2));
    _Float16*  rgbP  = (_Float16*) (ws + alloc((size_t)BATCH * 3 * HW0 * 2));
    _Float16*  rgbT  = (_Float16*) (ws + alloc((size_t)BATCH * 3 * HW0 * 2));
    char*      R1    =              ws + alloc(A1P * 2);                        // 135 MB
    char*      R2    =              ws + alloc(A1P * 2);                        // 135 MB
    _Float16*  fP    = (_Float16*) (ws + alloc(FNP * 2));
    _Float16*  fT    = (_Float16*) (ws + alloc(FNP * 2));

    float*    hbuf  = (float*)R1;      // SSIM moments (before convs touch R1)
    _Float16* a1p   = (_Float16*)R1;   // conv1 out, padded NHWC @512x64
    _Float16* a2p   = (_Float16*)R2;   // conv2 out, padded NHWC @512x64
    _Float16* plp   = (_Float16*)R1;   // pooled, padded NHWC @256x64 (a1 dead)
    _Float16* a3p   = (_Float16*)R2;   // conv3 out, padded NHWC @256x128 (a2 dead)

    // --- init + weight packing (tap-major K layout) ---
    init_acc_kernel<<<1, 32, 0, stream>>>(acc);
    pack_weights_kernel<<<(576  * 64  + 255) / 256, 256, 0, stream>>>(w2, p2, 64,  64);
    pack_weights_kernel<<<(576  * 128 + 255) / 256, 256, 0, stream>>>(w3, p3, 128, 64);
    pack_weights_kernel<<<(1152 * 128 + 255) / 256, 256, 0, stream>>>(w4, p4, 128, 128);

    // --- elementwise: LAB->RGB + MSE ---
    lab2rgb_mse_kernel<<<(BATCH * HW0) / 256, 256, 0, stream>>>(pred, targ, rgbP, rgbT, acc);

    // --- SSIM (uses R1 as scratch, before convs) ---
    ssim_h_kernel<<<(BATCH * 2 * HW0) / 256, 256, 0, stream>>>(pred, targ, hbuf);
    ssim_v_kernel<<<(BATCH * 2 * HW0) / 256, 256, 0, stream>>>(hbuf, acc);

    // --- VGG feature chains (pred then target, reusing R1/R2) ---
    const long inBS0  = (long)H0P * W0P * 64;    // padded @512 x 64ch
    const long inBS1  = (long)H1P * W1P * 64;    // padded @256 x 64ch
    const long inBS1c = (long)H1P * W1P * 128;   // padded @256 x 128ch
    const long fBS    = (long)HW1 * 128;

    const _Float16* rgbs[2]  = { rgbP, rgbT };
    _Float16*       fouts[2] = { fP, fT };
    for (int s = 0; s < 2; ++s) {
        // conv1: 3 -> 64 @512, relu (direct, writes a1p interior)
        zero_b128_kernel<<<(int)((A1P / 8 + 255) / 256), 256, 0, stream>>>((uint4*)a1p, (int)(A1P / 8));
        conv1_direct_kernel<<<dim3(HW0 / 4, BATCH), 256, 0, stream>>>(rgbs[s], w1, b1, a1p);
        // conv2: 64 -> 64 @512, relu
        zero_b128_kernel<<<(int)((A1P / 8 + 255) / 256), 256, 0, stream>>>((uint4*)a2p, (int)(A1P / 8));
        conv3x3_wmma_kernel<64><<<dim3(HW0 / 256, 64 / 32, BATCH), 256, 0, stream>>>(
            a1p, p2, b2, a2p + ((size_t)W0P + 1) * 64, 64, W0, W0P, W0P, inBS0, inBS0, 1);
        // maxpool 2x2 (a1 dead -> plp in R1)
        zero_b128_kernel<<<(int)((PLP / 8 + 255) / 256), 256, 0, stream>>>((uint4*)plp, (int)(PLP / 8));
        maxpool_kernel<<<(BATCH * HW1 * 64) / 256, 256, 0, stream>>>(a2p, plp);
        // conv3: 64 -> 128 @256, relu (a2 dead -> a3p in R2)
        zero_b128_kernel<<<(int)((A3P / 8 + 255) / 256), 256, 0, stream>>>((uint4*)a3p, (int)(A3P / 8));
        conv3x3_wmma_kernel<64><<<dim3(HW1 / 256, 128 / 32, BATCH), 256, 0, stream>>>(
            plp, p3, b3, a3p + ((size_t)W1P + 1) * 128, 128, W1, W1P, W1P, inBS1, inBS1c, 1);
        // conv4: 128 -> 128 @256, no relu, unpadded output
        conv3x3_wmma_kernel<128><<<dim3(HW1 / 256, 128 / 32, BATCH), 256, 0, stream>>>(
            a3p, p4, b4, fouts[s], 128, W1, W1P, W1, inBS1c, fBS, 0);
    }

    // --- perceptual reduction + combine ---
    perc_reduce_kernel<<<4096, 256, 0, stream>>>(fP, fT, acc, (int)FNP);
    combine_kernel<<<1, 32, 0, stream>>>(acc, out);
}
